// RRN_48146583388739
// MI455X (gfx1250) — matrix operand order repaired and checked by reading, not compiled, not run on hardware
//
#include <hip/hip_runtime.h>
#include <cstdint>
#include <cstddef>

// ---------------------------------------------------------------------------
// Types for CDNA5 WMMA (wave32): D(16x16 f32) = A(16x32 f16) * B(32x16 f16) + C
// ---------------------------------------------------------------------------
typedef __attribute__((ext_vector_type(16))) _Float16 v16h;
typedef __attribute__((ext_vector_type(8)))  float    v8f;
typedef _Float16 h16;

union HFrag {
    v16h     h;
    uint32_t u[8];   // 8 dwords = 16 halves
};

__device__ __forceinline__ v8f wmma_16x16x32(const HFrag& a, const HFrag& b, v8f c) {
    return __builtin_amdgcn_wmma_f32_16x16x32_f16(
        /*neg_a=*/false, a.h, /*neg_b=*/false, b.h,
        /*c_mod=*/(short)0, c, /*reuse_a=*/false, /*reuse_b=*/false);
}

__device__ __forceinline__ float sigf(float x) { return 1.0f / (1.0f + __expf(-x)); }

// Problem constants
#define BB 16
#define NN 128
#define HH 128
#define BN (BB * NN)     // 2048

// ---------------------------------------------------------------------------
// f32 -> f16 (strided sub-matrix) convert: dst[r*cols+c] = src[r*stride+coff+c]
// ---------------------------------------------------------------------------
__global__ void cvt_f32_to_f16(const float* __restrict__ src, h16* __restrict__ dst,
                               int rows, int cols, int stride, int coff) {
    int idx = blockIdx.x * blockDim.x + threadIdx.x;
    int tot = rows * cols;
    if (idx >= tot) return;
    int r = idx / cols;
    int c = idx - r * cols;
    dst[idx] = (h16)src[(size_t)r * stride + coff + c];
}

// ---------------------------------------------------------------------------
// Generic WMMA GEMM: C[M,N] = act( X[M,K] * W[N,K]^T + bias ), f16 in, f32 acc.
// 4 waves per block; each wave computes one 16x16 tile (flattened tile index).
// M,N,K multiples of 16; K%32==0.
// ---------------------------------------------------------------------------
__global__ __launch_bounds__(128)
void wmma_gemm(const h16* __restrict__ X, const h16* __restrict__ W,
               const float* __restrict__ bias,
               float* __restrict__ outF, h16* __restrict__ outH,
               int M, int N, int K, int relu) {
    const int lane  = threadIdx.x & 31;
    const int wv    = threadIdx.x >> 5;
    const int halfw = lane >> 4;
    const int lm    = lane & 15;

    const int Mt = M >> 4, Nt = N >> 4;
    const int t  = blockIdx.x * 4 + wv;
    if (t >= Mt * Nt) return;            // wave-uniform exit
    const int mt = t / Nt;
    const int nt = t - mt * Nt;

    const h16* xrow = X + (size_t)(mt * 16 + lm) * K;
    const h16* wrow = W + (size_t)(nt * 16 + lm) * K;

    v8f acc = {};
    for (int kk = 0; kk < K; kk += 32) {
        HFrag a, b;
#pragma unroll
        for (int r = 0; r < 8; ++r) {
            int ka = kk + 2 * r + ((r >= 4) ? 8 : 0) + 8 * halfw;  // A: M x K layout
            a.u[r] = *(const uint32_t*)(xrow + ka);
            int kb = kk + 2 * r + ((r >= 4) ? 8 : 0) + 8 * halfw;  // B: K x N layout
            b.u[r] = *(const uint32_t*)(wrow + kb);
        }
        acc = wmma_16x16x32(a, b, acc);
    }

    const int col = nt * 16 + lm;
    const float bv = bias ? bias[col] : 0.0f;
#pragma unroll
    for (int v = 0; v < 8; ++v) {
        int row = mt * 16 + v + 8 * halfw;
        float x = acc[v] + bv;
        if (relu) x = fmaxf(x, 0.0f);
        if (outF) outF[(size_t)row * N + col] = x;
        if (outH) outH[(size_t)row * N + col] = (h16)x;
    }
}

// ---------------------------------------------------------------------------
// Fused pairwise-message kernel. One workgroup (8 waves) per (b,i).
//   S0[j,:] = relu(Apre[b,j,:] + Bpre[b,i,:] + f_b0)        (128 x 256, LDS f16)
//   H1      = relu(S0 @ f_w1^T + f_b1)                      (128 x 256, LDS f16)
//   msg     = relu(H1 @ f_w2^T + f_b2)                      (128 x 128)
//   summ[b,i,:] = sum_j msg[j,:]
// Loop nest: each wave owns N-tiles and iterates M-tiles with 8 accumulators,
// so B-fragments (weights) are read from L2 exactly once per workgroup.
// ---------------------------------------------------------------------------
__global__ __launch_bounds__(256)
void pair_messages(const float* __restrict__ Apre,   // [B*N, 256]
                   const float* __restrict__ Bpre,   // [B*N, 256]
                   const float* __restrict__ fb0,    // [256]
                   const h16*  __restrict__ w1,      // [256, 256] f16
                   const float* __restrict__ fb1,    // [256]
                   const h16*  __restrict__ w2,      // [128, 256] f16
                   const float* __restrict__ fb2,    // [128]
                   float* __restrict__ summ)         // [B*N, 128]
{
    __shared__ h16 S0[128 * 256];   // 64 KB
    __shared__ h16 H1[128 * 256];   // 64 KB

    const int tid = threadIdx.x;
    const int bi  = blockIdx.x;          // b*128 + i
    const int b   = bi >> 7;

    const float* Ablk = Apre + (size_t)b * NN * 256;
    const float* Brow = Bpre + (size_t)bi * 256;

    // Stage layer-0 activations into LDS
    for (int idx = tid; idx < 128 * 256; idx += 256) {
        int j = idx >> 8;
        int o = idx & 255;
        float v = Ablk[(size_t)j * 256 + o] + Brow[o] + fb0[o];
        S0[idx] = (h16)fmaxf(v, 0.0f);
    }
    __syncthreads();

    const int wave  = tid >> 5;
    const int lane  = tid & 31;
    const int halfw = lane >> 4;
    const int lm    = lane & 15;

    // ---- Layer 1: H1 = relu(S0 @ w1^T + fb1).
    // Wave owns N-tiles {2*wave, 2*wave+1}; 8 accumulators over the M-tiles.
#pragma unroll 1
    for (int nn = 0; nn < 2; ++nn) {
        const int nt = wave * 2 + nn;
        const int n  = nt * 16 + lm;
        const h16* wrow = w1 + (size_t)n * 256;

        v8f acc[8];
#pragma unroll
        for (int m = 0; m < 8; ++m) acc[m] = (v8f){};

#pragma unroll 1
        for (int kk = 0; kk < 8; ++kk) {
            HFrag bf;
#pragma unroll
            for (int r = 0; r < 8; ++r)
                bf.u[r] = *(const uint32_t*)(wrow + kk * 32 + 2 * r + ((r >= 4) ? 8 : 0) + 8 * halfw);
#pragma unroll
            for (int m = 0; m < 8; ++m) {
                HFrag af;
#pragma unroll
                for (int r = 0; r < 8; ++r)
                    af.u[r] = *(const uint32_t*)(S0 + (m * 16 + lm) * 256 + kk * 32 +
                                                 2 * r + ((r >= 4) ? 8 : 0) + 8 * halfw);
                acc[m] = wmma_16x16x32(af, bf, acc[m]);
            }
        }

        const float bv = fb1[n];
#pragma unroll
        for (int m = 0; m < 8; ++m)
#pragma unroll
            for (int v = 0; v < 8; ++v)
                H1[(m * 16 + v + 8 * halfw) * 256 + n] = (h16)fmaxf(acc[m][v] + bv, 0.0f);
    }
    __syncthreads();

    // ---- Layer 2 + column-sum over j, fused. Wave owns N-tile = wave.
    {
        const int nt = wave;
        const int n  = nt * 16 + lm;
        const h16* wrow = w2 + (size_t)n * 256;

        v8f acc[8];
#pragma unroll
        for (int m = 0; m < 8; ++m) acc[m] = (v8f){};

#pragma unroll 1
        for (int kk = 0; kk < 8; ++kk) {
            HFrag bf;
#pragma unroll
            for (int r = 0; r < 8; ++r)
                bf.u[r] = *(const uint32_t*)(wrow + kk * 32 + 2 * r + ((r >= 4) ? 8 : 0) + 8 * halfw);
#pragma unroll
            for (int m = 0; m < 8; ++m) {
                HFrag af;
#pragma unroll
                for (int r = 0; r < 8; ++r)
                    af.u[r] = *(const uint32_t*)(H1 + (m * 16 + lm) * 256 + kk * 32 +
                                                 2 * r + ((r >= 4) ? 8 : 0) + 8 * halfw);
                acc[m] = wmma_16x16x32(af, bf, acc[m]);
            }
        }

        const float bv = fb2[n];
        float part = 0.0f;
#pragma unroll
        for (int m = 0; m < 8; ++m)
#pragma unroll
            for (int v = 0; v < 8; ++v)
                part += fmaxf(acc[m][v] + bv, 0.0f);

        // lanes L and L+16 cover complementary row-halves of column n
        part += __shfl_xor(part, 16, 32);
        if (lane < 16) summ[(size_t)bi * 128 + n] = part;
    }
}

// ---------------------------------------------------------------------------
// inpg[r, 0:128] = x[r], inpg[r, 128:256] = sum_messages[r]   (f16)
// ---------------------------------------------------------------------------
__global__ void concat_xg(const float* __restrict__ x, const float* __restrict__ summ,
                          h16* __restrict__ dst) {
    int idx = blockIdx.x * blockDim.x + threadIdx.x;
    if (idx >= BN * 256) return;
    int r = idx >> 8;
    int c = idx & 255;
    float v = (c < 128) ? x[(size_t)r * 128 + c] : summ[(size_t)r * 128 + (c - 128)];
    dst[idx] = (h16)v;
}

// ---------------------------------------------------------------------------
// LSTM single-step nonlinearity (pytorch gate order i,f,g,o)
// ---------------------------------------------------------------------------
__global__ void lstm_cell(const float* __restrict__ gA, const float* __restrict__ gB,
                          const float* __restrict__ bih, const float* __restrict__ bhh,
                          const float* __restrict__ c0,
                          float* __restrict__ out_g, float* __restrict__ h_n,
                          float* __restrict__ c_n) {
    int idx = blockIdx.x * blockDim.x + threadIdx.x;
    if (idx >= BN * HH) return;
    int r = idx >> 7;
    int c = idx & 127;
    const float* ga = gA + (size_t)r * 512;
    const float* gb = gB + (size_t)r * 512;
    float ig = ga[c]       + gb[c]       + bih[c]       + bhh[c];
    float fg = ga[c + 128] + gb[c + 128] + bih[c + 128] + bhh[c + 128];
    float gg = ga[c + 256] + gb[c + 256] + bih[c + 256] + bhh[c + 256];
    float og = ga[c + 384] + gb[c + 384] + bih[c + 384] + bhh[c + 384];
    float cv = sigf(fg) * c0[idx] + sigf(ig) * tanhf(gg);
    float hv = sigf(og) * tanhf(cv);
    out_g[idx] = hv;
    h_n[idx]   = hv;
    c_n[idx]   = cv;
}

// ---------------------------------------------------------------------------
// sum over n: sumh[b, :] = sum_n out_g[b, n, :]   (emit f16 for the o-MLP)
// ---------------------------------------------------------------------------
__global__ void sum_over_n(const float* __restrict__ hn, h16* __restrict__ outh) {
    int b = blockIdx.x;      // 16
    int t = threadIdx.x;     // 128
    float acc = 0.0f;
    for (int n = 0; n < NN; ++n) acc += hn[((size_t)b * NN + n) * HH + t];
    outh[b * HH + t] = (h16)acc;
}

// ---------------------------------------------------------------------------
// Launcher
// ---------------------------------------------------------------------------
extern "C" void kernel_launch(void* const* d_in, const int* in_sizes, int n_in,
                              void* d_out, int out_size, void* d_ws, size_t ws_size,
                              hipStream_t stream) {
    (void)in_sizes; (void)n_in; (void)out_size; (void)ws_size;

    const float* x        = (const float*)d_in[0];
    const float* hidden   = (const float*)d_in[1];
    const float* h0       = (const float*)d_in[2];
    const float* c0       = (const float*)d_in[3];
    const float* f_w0     = (const float*)d_in[4];
    const float* f_b0     = (const float*)d_in[5];
    const float* f_w1     = (const float*)d_in[6];
    const float* f_b1     = (const float*)d_in[7];
    const float* f_w2     = (const float*)d_in[8];
    const float* f_b2     = (const float*)d_in[9];
    const float* g_w0     = (const float*)d_in[10];
    const float* g_b0     = (const float*)d_in[11];
    const float* g_w1     = (const float*)d_in[12];
    const float* g_b1     = (const float*)d_in[13];
    const float* g_w2     = (const float*)d_in[14];
    const float* g_b2     = (const float*)d_in[15];
    const float* lstm_wih = (const float*)d_in[16];
    const float* lstm_whh = (const float*)d_in[17];
    const float* lstm_bih = (const float*)d_in[18];
    const float* lstm_bhh = (const float*)d_in[19];
    const float* o_w0     = (const float*)d_in[20];
    const float* o_b0     = (const float*)d_in[21];
    const float* o_w1     = (const float*)d_in[22];
    const float* o_b1     = (const float*)d_in[23];

    float* out_final = (float*)d_out;                       // [16, 64]
    float* out_g     = out_final + 16 * 64;                 // [16,128,128]
    float* h_n       = out_g + BN * HH;                     // [1,2048,128]
    float* c_n       = h_n + BN * HH;                       // [1,2048,128]

    // Workspace carve-out
    char*  wsb = (char*)d_ws;
    size_t off = 0;
    auto take = [&](size_t bytes) -> void* {
        void* p = wsb + off;
        off = (off + bytes + 255) & ~(size_t)255;
        return p;
    };
    h16*  hid_h  = (h16*)take((size_t)BN * 128 * 2);
    h16*  h0_h   = (h16*)take((size_t)BN * 128 * 2);
    h16*  fw0a_h = (h16*)take(256 * 128 * 2);
    h16*  fw0b_h = (h16*)take(256 * 128 * 2);
    h16*  fw1_h  = (h16*)take(256 * 256 * 2);
    h16*  fw2_h  = (h16*)take(128 * 256 * 2);
    h16*  gw0_h  = (h16*)take(256 * 256 * 2);
    h16*  gw1_h  = (h16*)take(256 * 256 * 2);
    h16*  gw2_h  = (h16*)take(128 * 256 * 2);
    h16*  wih_h  = (h16*)take(512 * 128 * 2);
    h16*  whh_h  = (h16*)take(512 * 128 * 2);
    h16*  ow0_h  = (h16*)take(256 * 128 * 2);
    h16*  ow1_h  = (h16*)take(64 * 256 * 2);
    h16*  inpg_h = (h16*)take((size_t)BN * 256 * 2);
    h16*  g0_h   = (h16*)take((size_t)BN * 256 * 2);
    h16*  g1_h   = (h16*)take((size_t)BN * 256 * 2);
    h16*  g2_h   = (h16*)take((size_t)BN * 128 * 2);
    h16*  sumh_h = (h16*)take(16 * 128 * 2);
    h16*  o0_h   = (h16*)take(16 * 256 * 2);
    float* Apre  = (float*)take((size_t)BN * 256 * 4);
    float* Bpre  = (float*)take((size_t)BN * 256 * 4);
    float* summ  = (float*)take((size_t)BN * 128 * 4);
    float* gatesA = (float*)take((size_t)BN * 512 * 4);
    float* gatesB = (float*)take((size_t)BN * 512 * 4);

    auto cvt = [&](const float* s, h16* d, int rows, int cols, int stride, int coff) {
        int tot = rows * cols;
        cvt_f32_to_f16<<<(tot + 255) / 256, 256, 0, stream>>>(s, d, rows, cols, stride, coff);
    };
    auto gemm = [&](const h16* X, const h16* W, const float* bias,
                    float* oF, h16* oH, int M, int N, int K, int relu) {
        int tiles = (M / 16) * (N / 16);
        wmma_gemm<<<(tiles + 3) / 4, 128, 0, stream>>>(X, W, bias, oF, oH, M, N, K, relu);
    };

    // --- Prep: f16 copies of activations/weights (f_w0 split into two [256,128]) ---
    cvt(hidden,   hid_h,  BN, 128, 128, 0);
    cvt(h0,       h0_h,   BN, 128, 128, 0);
    cvt(f_w0,     fw0a_h, 256, 128, 256, 0);
    cvt(f_w0,     fw0b_h, 256, 128, 256, 128);
    cvt(f_w1,     fw1_h,  256, 256, 256, 0);
    cvt(f_w2,     fw2_h,  128, 256, 256, 0);
    cvt(g_w0,     gw0_h,  256, 256, 256, 0);
    cvt(g_w1,     gw1_h,  256, 256, 256, 0);
    cvt(g_w2,     gw2_h,  128, 256, 256, 0);
    cvt(lstm_wih, wih_h,  512, 128, 128, 0);
    cvt(lstm_whh, whh_h,  512, 128, 128, 0);
    cvt(o_w0,     ow0_h,  256, 128, 128, 0);
    cvt(o_w1,     ow1_h,  64,  256, 256, 0);

    // --- f-MLP layer 0 decomposition: Apre = hid @ W0a^T ; Bpre = hid @ W0b^T ---
    gemm(hid_h, fw0a_h, nullptr, Apre, nullptr, BN, 256, 128, 0);
    gemm(hid_h, fw0b_h, nullptr, Bpre, nullptr, BN, 256, 128, 0);

    // --- Fused pairwise messages + sum over j (the 51.5 GFLOP hot loop) ---
    pair_messages<<<BN, 256, 0, stream>>>(Apre, Bpre, f_b0, fw1_h, f_b1, fw2_h, f_b2, summ);

    // --- g-MLP ---
    concat_xg<<<(BN * 256 + 255) / 256, 256, 0, stream>>>(x, summ, inpg_h);
    gemm(inpg_h, gw0_h, g_b0, nullptr, g0_h, BN, 256, 256, 1);
    gemm(g0_h,   gw1_h, g_b1, nullptr, g1_h, BN, 256, 256, 1);
    gemm(g1_h,   gw2_h, g_b2, nullptr, g2_h, BN, 128, 256, 1);

    // --- LSTM step ---
    gemm(g2_h, wih_h, nullptr, gatesA, nullptr, BN, 512, 128, 0);
    gemm(h0_h, whh_h, nullptr, gatesB, nullptr, BN, 512, 128, 0);
    lstm_cell<<<(BN * HH + 255) / 256, 256, 0, stream>>>(
        gatesA, gatesB, lstm_bih, lstm_bhh, c0, out_g, h_n, c_n);

    // --- Output MLP ---
    sum_over_n<<<16, 128, 0, stream>>>(h_n, sumh_h);
    gemm(sumh_h, ow0_h, o_b0, nullptr, o0_h, 16, 256, 128, 1);
    gemm(o0_h,   ow1_h, o_b1, out_final, nullptr, 16, 64, 256, 1);
}
